// MulLayer_48687749267524
// MI455X (gfx1250) — compile-verified, matrix operand order checked by reading, not run on hardware
//
#include <hip/hip_runtime.h>
#include <hip/hip_bf16.h>

typedef __attribute__((ext_vector_type(16))) _Float16 v16h;
typedef __attribute__((ext_vector_type(8)))  float    v8f;
typedef __attribute__((ext_vector_type(2)))  float    v2f;
typedef __attribute__((vector_size(16)))     int      ai4;   // 4 x i32, matches async-LDS builtin pointee

#define MS    64
#define HW    16384      // 128*128
#define KSEL  8192
#define POOLW 256

// GEMM macro-tile (per 256-thread block = 8 waves, 2x4 wave grid of 16x16 tiles)
#define BM 32
#define BN 64
#define BK 32

union V16H { v16h v; _Float16 e[16]; };
union V8F  { v8f  v; float    e[8];  };

// ---- gfx1250 async global->LDS staging (ASYNCcnt path), with safe fallback ----
#if __has_builtin(__builtin_amdgcn_global_load_async_to_lds_b128) && \
    __has_builtin(__builtin_amdgcn_s_wait_asynccnt)
#define USE_ASYNC_LDS 1
#else
#define USE_ASYNC_LDS 0
#endif

__device__ __forceinline__ void stage16_to_lds(const float* g, float* l)
{
#if USE_ASYNC_LDS
    __builtin_amdgcn_global_load_async_to_lds_b128(
        (__attribute__((address_space(1))) ai4*)g,
        (__attribute__((address_space(3))) ai4*)l, 0, 0);
#else
    l[0] = g[0]; l[1] = g[1]; l[2] = g[2]; l[3] = g[3];
#endif
}

__device__ __forceinline__ void stage_wait()
{
#if USE_ASYNC_LDS
    __builtin_amdgcn_s_wait_asynccnt(0);
#endif
}

// ---------------- generic 3x3 stride-2 pad-1 conv (NCHW, N=1) ----------------
__global__ void conv3x3s2_kernel(const float* __restrict__ in,
                                 const float* __restrict__ w,
                                 const float* __restrict__ bias,
                                 float* __restrict__ out,
                                 int Cin, int Cout, int Hin, int Win,
                                 int Hout, int Wout, int relu)
{
    int idx = blockIdx.x * blockDim.x + threadIdx.x;
    int total = Cout * Hout * Wout;
    if (idx >= total) return;
    int ox = idx % Wout;
    int oy = (idx / Wout) % Hout;
    int oc = idx / (Wout * Hout);
    float acc = bias[oc];
    int iy0 = oy * 2 - 1;
    int ix0 = ox * 2 - 1;
    for (int c = 0; c < Cin; ++c) {
        const float* inC = in + (size_t)c * Hin * Win;
        const float* wC  = w + ((size_t)oc * Cin + c) * 9;
        for (int ky = 0; ky < 3; ++ky) {
            int iy = iy0 + ky;
            if (iy < 0 || iy >= Hin) continue;
            for (int kx = 0; kx < 3; ++kx) {
                int ix = ix0 + kx;
                if (ix < 0 || ix >= Win) continue;
                acc = fmaf(inC[iy * Win + ix], wC[ky * 3 + kx], acc);
            }
        }
    }
    if (relu) acc = fmaxf(acc, 0.0f);
    out[idx] = acc;
}

// ---------------- gather + max-pool ----------------
__global__ void pool_gather_kernel(const float* __restrict__ feat,
                                   const int* __restrict__ fg,
                                   float* __restrict__ pooled)
{
    int idx = blockIdx.x * blockDim.x + threadIdx.x;   // 0..4095
    if (idx >= 16 * POOLW) return;
    int p = idx % POOLW;
    int c = idx / POOLW;
    const int g = KSEL / POOLW;                        // 32
    float m = -3.402823466e38f;
    for (int j = 0; j < g; ++j) {
        int col = fg[p * g + j];
        m = fmaxf(m, feat[c * HW + col]);
    }
    pooled[c * POOLW + p] = m;
}

// ---------------- fc GEMV: y = pooled(4096) @ fcw^T + fcb ----------------
__global__ void fc_kernel(const float* __restrict__ pooled,
                          const float* __restrict__ fcw,
                          const float* __restrict__ fcb,
                          float* __restrict__ outMat)
{
    int j = blockIdx.x * blockDim.x + threadIdx.x;
    if (j >= MS * MS) return;
    const float* wr = fcw + (size_t)j * (16 * POOLW);
    float acc = fcb[j];
    for (int i = 0; i < 16 * POOLW; ++i) {
        if ((i & 255) == 0) __builtin_prefetch(wr + i + 256, 0, 1);
        acc = fmaf(pooled[i], wr[i], acc);
    }
    outMat[j] = acc;
}

// ---------------- T = sMat @ cMat via full-precision V_WMMA_F32_16X16X4_F32 ----
// 64x64x64: 16 tiles, one wave each -> launch <<<2,256>>>.
__global__ void matmulT_wmma(const float* __restrict__ A,
                             const float* __restrict__ B,
                             float* __restrict__ C)
{
    int wave  = (blockIdx.x * blockDim.x + threadIdx.x) >> 5;
    int lane  = threadIdx.x & 31;
    int mTile = wave >> 2;
    int nTile = wave & 3;
    int row15 = lane & 15;
    int hi    = (lane >= 16) ? 1 : 0;
    int hi8   = hi * 8;
    V8F r;
#if __has_builtin(__builtin_amdgcn_wmma_f32_16x16x4_f32)
    // A 16x4: lane m=lane&15, VGPR j holds K = 2*hi + j ; B 4x16 symmetric.
    int kBase = hi * 2;
    v8f acc = {};
    for (int k0 = 0; k0 < MS; k0 += 4) {
        v2f a, b;
        a.x = A[(mTile * 16 + row15) * MS + k0 + kBase];
        a.y = A[(mTile * 16 + row15) * MS + k0 + kBase + 1];
        b.x = B[(k0 + kBase) * MS + nTile * 16 + row15];
        b.y = B[(k0 + kBase + 1) * MS + nTile * 16 + row15];
        acc = __builtin_amdgcn_wmma_f32_16x16x4_f32(false, a, false, b,
                                                    (short)0, acc, false, false);
    }
    r.v = acc;
#else
    // scalar fallback producing the same C/D lane layout
    for (int rr = 0; rr < 8; ++rr) {
        int mo = mTile * 16 + rr + hi8;
        int n  = nTile * 16 + row15;
        float s = 0.f;
        for (int k = 0; k < MS; ++k) s = fmaf(A[mo * MS + k], B[k * MS + n], s);
        r.e[rr] = s;
    }
#endif
#pragma unroll
    for (int rr = 0; rr < 8; ++rr) {
        int mo = mTile * 16 + rr + hi8;
        C[mo * MS + nTile * 16 + row15] = r.e[rr];
    }
}

// ---------------- LDS-staged WMMA GEMM: C(MxN) = A(MxK) x B(KxN) (+bias[row]) ----
// Block = 256 threads (8 waves) computes a BMxBN (32x64) macro tile.
// Each K-chunk (32) is staged to LDS with async b128 global->LDS loads,
// s_wait_asynccnt + barrier, then waves build f16 fragments from LDS and
// issue v_wmma_f32_16x16x32_f16.
__global__ void wmma_gemm_lds(const float* __restrict__ A, int lda,
                              const float* __restrict__ B, int ldb,
                              const float* __restrict__ bias,
                              float* __restrict__ C, int ldc,
                              int M, int N, int K)
{
    __shared__ float lA[BM * BK];   // 32x32
    __shared__ float lB[BK * BN];   // 32x64

    int nBlocks = N / BN;
    int mBase = (blockIdx.x / nBlocks) * BM;
    int nBase = (blockIdx.x % nBlocks) * BN;

    int tid   = threadIdx.x;
    int waveI = tid >> 5;
    int lane  = tid & 31;
    int mt    = waveI >> 2;            // 0..1
    int nt    = waveI & 3;             // 0..3
    int row15 = lane & 15;
    int hi8   = (lane >= 16) ? 8 : 0;

    v8f acc = {};
    for (int k0 = 0; k0 < K; k0 += BK) {
        // ---- stage A chunk: 32x32 floats = 256 quads, one per thread ----
        {
            int r  = tid >> 3;             // 0..31
            int c4 = (tid & 7) << 2;       // 0,4,...,28
            stage16_to_lds(&A[(size_t)(mBase + r) * lda + k0 + c4], &lA[r * BK + c4]);
        }
        // ---- stage B chunk: 32x64 floats = 512 quads, two per thread ----
#pragma unroll
        for (int q = 0; q < 2; ++q) {
            int i  = tid + q * 256;        // quad index 0..511
            int k  = i >> 4;               // 0..31
            int n4 = (i & 15) << 2;        // 0,4,...,60
            stage16_to_lds(&B[(size_t)(k0 + k) * ldb + nBase + n4], &lB[k * BN + n4]);
        }
        stage_wait();
        __syncthreads();

        // ---- build f16 fragments from LDS, per CDNA5 A 16x32 / B 32x16 layout ----
        V16H a, b;
#pragma unroll
        for (int h = 0; h < 16; ++h) {
            int kl = ((h >> 3) << 4) + hi8 + (h & 7);          // local k 0..31
            a.e[h] = (_Float16)lA[(mt * 16 + row15) * BK + kl];
            b.e[h] = (_Float16)lB[kl * BN + nt * 16 + row15];
        }
        acc = __builtin_amdgcn_wmma_f32_16x16x32_f16(false, a.v, false, b.v,
                                                     (short)0, acc, false, false);
        __syncthreads();   // protect LDS before next chunk's staging
    }

    V8F r; r.v = acc;
    int n = nBase + nt * 16 + row15;
    if (bias != nullptr) {
#pragma unroll
        for (int rr = 0; rr < 8; ++rr) {
            int mo = mBase + mt * 16 + rr + hi8;
            C[(size_t)mo * ldc + n] = r.e[rr] + bias[mo];
        }
    } else {
#pragma unroll
        for (int rr = 0; rr < 8; ++rr) {
            int mo = mBase + mt * 16 + rr + hi8;
            C[(size_t)mo * ldc + n] = r.e[rr];
        }
    }
}

// ---------------- direct-register WMMA GEMM with gathered B columns ----------------
// Used for transFG = T(64x64) @ ccf[:, fgc](64x8192); gather defeats LDS reuse.
__global__ void wmma_gemm_gather(const float* __restrict__ A, int lda,
                                 const float* __restrict__ B, int ldb,
                                 const int* __restrict__ bcol_idx,
                                 float* __restrict__ C, int ldc,
                                 int M, int N, int K)
{
    int wave = (blockIdx.x * blockDim.x + threadIdx.x) >> 5;
    int lane = threadIdx.x & 31;
    int nTiles = N >> 4;
    int mTile = wave / nTiles;
    int nTile = wave % nTiles;
    if (mTile >= (M >> 4)) return;                 // wave-uniform (never trips)
    int row15 = lane & 15;
    int hi8   = (lane >= 16) ? 8 : 0;

    int m = mTile * 16 + row15;
    int n = nTile * 16 + row15;
    int bcol = bcol_idx[n];

    v8f acc = {};
    for (int k0 = 0; k0 < K; k0 += 32) {
        V16H a, b;
#pragma unroll
        for (int h = 0; h < 16; ++h) {
            int k = k0 + ((h >> 3) << 4) + hi8 + (h & 7);
            a.e[h] = (_Float16)A[(size_t)m * lda + k];
            b.e[h] = (_Float16)B[(size_t)k * ldb + bcol];
        }
        acc = __builtin_amdgcn_wmma_f32_16x16x32_f16(false, a.v, false, b.v,
                                                     (short)0, acc, false, false);
    }
    V8F r; r.v = acc;
#pragma unroll
    for (int rr = 0; rr < 8; ++rr) {
        int mo = mTile * 16 + rr + hi8;
        C[(size_t)mo * ldc + n] = r.e[rr];
    }
}

// ---------------- per-row mean subtraction (64 rows x 16384) ----------------
__global__ void mean_sub_kernel(float* __restrict__ ccf)
{
    __shared__ float red[256];
    int row = blockIdx.x;
    float s = 0.f;
    for (int i = threadIdx.x; i < HW; i += blockDim.x)
        s += ccf[(size_t)row * HW + i];
    red[threadIdx.x] = s;
    __syncthreads();
    for (int off = 128; off > 0; off >>= 1) {
        if ((int)threadIdx.x < off) red[threadIdx.x] += red[threadIdx.x + off];
        __syncthreads();
    }
    float mean = red[0] / (float)HW;
    for (int i = threadIdx.x; i < HW; i += blockDim.x)
        ccf[(size_t)row * HW + i] -= mean;
}

// ---------------- deterministic scatter (sorted mask -> highest dup wins) ----------------
__global__ void scatter_kernel(const float* __restrict__ transFG,
                               const int* __restrict__ fgc,
                               float* __restrict__ ccf)
{
    int k = blockIdx.x * blockDim.x + threadIdx.x;
    if (k >= KSEL) return;
    int idx = fgc[k];
    bool winner = (k == KSEL - 1) || (fgc[k + 1] != idx);
    if (!winner) return;
    for (int m2 = 0; m2 < MS; ++m2)
        ccf[(size_t)m2 * HW + idx] = transFG[(size_t)m2 * KSEL + k];
}

extern "C" void kernel_launch(void* const* d_in, const int* in_sizes, int n_in,
                              void* d_out, int out_size, void* d_ws, size_t ws_size,
                              hipStream_t stream)
{
    (void)in_sizes; (void)n_in; (void)out_size; (void)ws_size;

    const float* cF      = (const float*)d_in[0];
    /* d_in[1] (sF) is unused by the reference */
    const float* content = (const float*)d_in[2];
    const float* style   = (const float*)d_in[3];
    const float* s_c1w = (const float*)d_in[4];
    const float* s_c1b = (const float*)d_in[5];
    const float* s_c2w = (const float*)d_in[6];
    const float* s_c2b = (const float*)d_in[7];
    const float* s_c3w = (const float*)d_in[8];
    const float* s_c3b = (const float*)d_in[9];
    const float* s_fcw = (const float*)d_in[10];
    const float* s_fcb = (const float*)d_in[11];
    const float* c_c1w = (const float*)d_in[12];
    const float* c_c1b = (const float*)d_in[13];
    const float* c_c2w = (const float*)d_in[14];
    const float* c_c2b = (const float*)d_in[15];
    const float* c_c3w = (const float*)d_in[16];
    const float* c_c3b = (const float*)d_in[17];
    const float* c_fcw = (const float*)d_in[18];
    const float* c_fcb = (const float*)d_in[19];
    const float* compress_w = (const float*)d_in[20];
    const float* compress_b = (const float*)d_in[21];
    const float* unzip_w    = (const float*)d_in[22];
    const float* unzip_b    = (const float*)d_in[23];
    const int*   fgc = (const int*)d_in[24];
    const int*   fgs = (const int*)d_in[25];

    // -------- workspace layout (floats) --------
    float* conv1buf = (float*)d_ws;                  // 64*512*512 = 16,777,216
    float* conv2buf = conv1buf + 16777216;           // 32*256*256 =  2,097,152
    float* conv3buf = conv2buf + 2097152;            // 16*128*128 =    262,144
    float* pooled   = conv3buf + 262144;             //                  4,096
    float* sMat     = pooled   + 4096;               //                  4,096
    float* cMat     = sMat     + 4096;               //                  4,096
    float* Tm       = cMat     + 4096;               //                  4,096
    float* ccf      = Tm       + 4096;               // 64*16384  =  1,048,576
    float* trans    = ccf      + 1048576;            // 64*8192   =    524,288

    auto runNet = [&](const float* img,
                      const float* c1w, const float* c1b,
                      const float* c2w, const float* c2b,
                      const float* c3w, const float* c3b,
                      const float* fcw, const float* fcb,
                      const int* fg, float* outMat) {
        conv3x3s2_kernel<<<65536, 256, 0, stream>>>(img,      c1w, c1b, conv1buf, 3,  64, 1024, 1024, 512, 512, 1);
        conv3x3s2_kernel<<<8192,  256, 0, stream>>>(conv1buf, c2w, c2b, conv2buf, 64, 32,  512,  512, 256, 256, 1);
        conv3x3s2_kernel<<<1024,  256, 0, stream>>>(conv2buf, c3w, c3b, conv3buf, 32, 16,  256,  256, 128, 128, 0);
        pool_gather_kernel<<<16, 256, 0, stream>>>(conv3buf, fg, pooled);
        fc_kernel<<<16, 256, 0, stream>>>(pooled, fcw, fcb, outMat);
    };

    runNet(style,   s_c1w, s_c1b, s_c2w, s_c2b, s_c3w, s_c3b, s_fcw, s_fcb, fgs, sMat);
    runNet(content, c_c1w, c_c1b, c_c2w, c_c2b, c_c3w, c_c3b, c_fcw, c_fcb, fgc, cMat);

    // T = sMat @ cMat (64x64x64) on the f32 matrix pipe
    matmulT_wmma<<<2, 256, 0, stream>>>(sMat, cMat, Tm);

    // ccf = compress_w(64x512) @ cF(512x16384) + compress_b
    // macro tiles: (64/32)*(16384/64) = 512 blocks
    wmma_gemm_lds<<<512, 256, 0, stream>>>(compress_w, 512, cF, HW,
                                           compress_b, ccf, HW, MS, HW, 512);

    // subtract per-row mean
    mean_sub_kernel<<<MS, 256, 0, stream>>>(ccf);

    // transFG = T(64x64) @ ccf[:, fgc](64x8192) ; 2048 tiles -> 256 blocks
    wmma_gemm_gather<<<256, 256, 0, stream>>>(Tm, MS, ccf, HW, fgc,
                                              trans, KSEL, MS, KSEL, MS);

    // ccf[:, fgc] = transFG (deterministic, last duplicate wins)
    scatter_kernel<<<KSEL / 256, 256, 0, stream>>>(trans, fgc, ccf);

    // out = unzip_w(512x64) @ ccf(64x16384) + unzip_b
    // macro tiles: (512/32)*(16384/64) = 4096 blocks
    wmma_gemm_lds<<<4096, 256, 0, stream>>>(unzip_w, MS, ccf, HW,
                                            unzip_b, (float*)d_out, HW, 512, HW, MS);
}
